// MultiLayerDH_SFNN_86474871538207
// MI455X (gfx1250) — matrix operand order repaired and checked by reading, not compiled, not run on hardware
//
#include <hip/hip_runtime.h>
#include <hip/hip_bf16.h>

typedef __attribute__((ext_vector_type(16))) _Float16 v16h;
typedef __attribute__((ext_vector_type(8)))  float    v8f;

#define B_  64
#define T_  1024
#define I_  256
#define H1_ 512
#define H2_ 512
#define O_  128

// A-matrix (16x32 f16) K index held by lane-group g (lane/16), half-element e (0..15).
// Per ISA 7.12.2: lanes0-15 V0..3 -> K0..7, V4..7 -> K16..23; lanes16-31 shifted by +8.
__device__ __forceinline__ int kmapA(int g, int e) {
  int v = e >> 1, p = e & 1;
  int base = (v < 4) ? (2 * v + p) : (16 + 2 * (v - 4) + p);
  return base + 8 * g;
}
// B-matrix (32x16 f16): same K order as A: lanes0-15 e0..7 -> K0..7, e8..15 -> K16..23; lanes16-31 shifted by +8.
__device__ __forceinline__ int kmapB(int g, int e) { return (e < 8) ? (e + 8 * g) : (8 + e + 8 * g); }

__device__ __forceinline__ float sigmoidf_(float v) { return 1.0f / (1.0f + __expf(-v)); }

// Async global->LDS copy of 16 bytes per lane (tracked by ASYNCcnt).
__device__ __forceinline__ void async_copy16(unsigned lds_off, unsigned long long gaddr) {
  asm volatile("global_load_async_to_lds_b128 %0, %1, off"
               :: "v"(lds_off), "v"(gaddr) : "memory");
}
__device__ __forceinline__ void wait_async0() {
  asm volatile("s_wait_asynccnt 0x0" ::: "memory");
}

// ---------------------------------------------------------------------------
// Pack x[B,T,I] (f32) -> f16 A-fragment order, M flattened as (t*64 + b).
// ---------------------------------------------------------------------------
__global__ __launch_bounds__(256) void pack_x_kernel(const float* __restrict__ x,
                                                     _Float16* __restrict__ xp) {
  int idx = blockIdx.x * 256 + threadIdx.x;           // 16,777,216 total
  int e  = idx & 15;
  int l  = (idx >> 4) & 31;
  int kt = (idx >> 9) & 7;
  int mt = idx >> 12;
  int m  = mt * 16 + (l & 15);
  int t  = m >> 6;
  int b  = m & 63;
  int k  = kt * 32 + kmapA(l >> 4, e);
  xp[idx] = (_Float16)x[((size_t)b * T_ + t) * I_ + k];
}

// ---------------------------------------------------------------------------
// Pack W1[512,256], W2[512,512], Wo[128,512] -> f16 B-fragment order.
// ---------------------------------------------------------------------------
__global__ __launch_bounds__(256) void pack_w_kernel(const float* __restrict__ W1,
                                                     const float* __restrict__ W2,
                                                     const float* __restrict__ Wo,
                                                     _Float16* __restrict__ w1p,
                                                     _Float16* __restrict__ w2p,
                                                     _Float16* __restrict__ wop) {
  int idx = blockIdx.x * 256 + threadIdx.x;
  if (idx < 131072) {                                  // W1: NT=32, KT=8
    int e = idx & 15, l = (idx >> 4) & 31, kt = (idx >> 9) & 7, nt = idx >> 12;
    int n = nt * 16 + (l & 15);
    int k = kt * 32 + kmapB(l >> 4, e);
    w1p[idx] = (_Float16)W1[n * I_ + k];
  } else if (idx < 131072 + 262144) {                  // W2: NT=32, KT=16
    int j = idx - 131072;
    int e = j & 15, l = (j >> 4) & 31, kt = (j >> 9) & 15, nt = j >> 13;
    int n = nt * 16 + (l & 15);
    int k = kt * 32 + kmapB(l >> 4, e);
    w2p[j] = (_Float16)W2[n * H1_ + k];
  } else if (idx < 131072 + 262144 + 65536) {          // Wo: NT=8, KT=16
    int j = idx - 131072 - 262144;
    int e = j & 15, l = (j >> 4) & 31, kt = (j >> 9) & 15, nt = j >> 13;
    int n = nt * 16 + (l & 15);
    int k = kt * 32 + kmapB(l >> 4, e);
    wop[j] = (_Float16)Wo[n * H2_ + k];
  }
}

// ---------------------------------------------------------------------------
// GEMM1: din1[M=65536][H1] = xp @ W1^T + b1 (f32 out). One block = 64 M-rows,
// wave w covers N columns [w*64, w*64+64). 4x4 WMMA tiles per wave.
// ---------------------------------------------------------------------------
__global__ __launch_bounds__(256) void gemm1_kernel(const _Float16* __restrict__ xp,
                                                    const _Float16* __restrict__ w1p,
                                                    const float* __restrict__ b1,
                                                    float* __restrict__ din1) {
  int w    = threadIdx.x >> 5;
  int lane = threadIdx.x & 31;
  int mblk = blockIdx.x;                                // 64 rows
  v8f zero = {0.f, 0.f, 0.f, 0.f, 0.f, 0.f, 0.f, 0.f};
  v8f c[4][4];
#pragma unroll
  for (int im = 0; im < 4; ++im)
#pragma unroll
    for (int jn = 0; jn < 4; ++jn) c[im][jn] = zero;

  int mt0 = mblk * 4;
  for (int kt = 0; kt < 8; ++kt) {
    v16h a[4], bfr[4];
#pragma unroll
    for (int im = 0; im < 4; ++im)
      a[im] = *(const v16h*)(xp + ((size_t)((mt0 + im) * 8 + kt) * 32 + lane) * 16);
#pragma unroll
    for (int jn = 0; jn < 4; ++jn)
      bfr[jn] = *(const v16h*)(w1p + ((size_t)((w * 4 + jn) * 8 + kt) * 32 + lane) * 16);
#pragma unroll
    for (int im = 0; im < 4; ++im)
#pragma unroll
      for (int jn = 0; jn < 4; ++jn)
        c[im][jn] = __builtin_amdgcn_wmma_f32_16x16x32_f16(
            false, a[im], false, bfr[jn], (short)0, c[im][jn], false, false);
  }

  int lg = lane >> 4, ln = lane & 15;
#pragma unroll
  for (int jn = 0; jn < 4; ++jn) {
    int n = (w * 4 + jn) * 16 + ln;
    float bias = b1[n];
#pragma unroll
    for (int im = 0; im < 4; ++im) {
      int mbase = mblk * 64 + im * 16 + 8 * lg;
#pragma unroll
      for (int r = 0; r < 8; ++r)
        din1[(size_t)(mbase + r) * H1_ + n] = c[im][jn][r] + bias;
    }
  }
}

// ---------------------------------------------------------------------------
// Serial scan: 4 blocks x 256 threads; block bg owns batch rows [bg*16, bg*16+16).
// Recurrent state in registers; spikes cross waves via LDS stored directly in
// WMMA A-fragment order; din1 slices double-buffered in LDS via async copies.
// ---------------------------------------------------------------------------
__global__ __launch_bounds__(256) void scan_kernel(const float* __restrict__ din1,
                                                   const _Float16* __restrict__ w2p,
                                                   const _Float16* __restrict__ wop,
                                                   const float* __restrict__ b2,
                                                   const float* __restrict__ bo,
                                                   const float* __restrict__ tau_m1,
                                                   const float* __restrict__ tau_n1,
                                                   const float* __restrict__ tau_m2,
                                                   const float* __restrict__ tau_n2,
                                                   float* __restrict__ out) {
  // Spikes in packed A-fragment order: [(kt*32 + laneRow)*16 + e]
  __shared__ __align__(32) _Float16 s1p[16 * H1_];
  __shared__ __align__(32) _Float16 s2p[16 * H2_];
  __shared__ __align__(16) float dbuf[2][16 * H1_];    // din1 slice double buffer
  __shared__ float beta1s[H1_], alpha1s[H1_];

  int tid = threadIdx.x, lane = tid & 31, w = tid >> 5;
  int lg = lane >> 4, ln = lane & 15;
  int b0 = blockIdx.x * 16;

  for (int h = tid; h < H1_; h += 256) {
    beta1s[h]  = sigmoidf_(tau_n1[h]);
    alpha1s[h] = sigmoidf_(tau_m1[h]);
  }

  // layer-2 per-owned-column params (wave w owns N-tiles w*4..w*4+3)
  float beta2r[4], alpha2r[4], b2r[4];
#pragma unroll
  for (int jj = 0; jj < 4; ++jj) {
    int n = (w * 4 + jj) * 16 + ln;
    beta2r[jj]  = sigmoidf_(tau_n2[n]);
    alpha2r[jj] = sigmoidf_(tau_m2[n]);
    b2r[jj]     = b2[n];
  }
  int   nO   = w * 16 + ln;      // output N-tile = wave id
  float bo_r = bo[nO];

  // layer-1 ownership: thread owns row m1, 32 contiguous h (one kt block)
  int m1 = tid >> 4;
  int hbase = (tid & 15) * 32;
  int kt1 = hbase >> 5;          // fixed K-tile of this thread's h range

  float mem1[32], d1[32];
  float mem2[4][8], d2[4][8];
#pragma unroll
  for (int i = 0; i < 32; ++i) { mem1[i] = 0.f; d1[i] = 0.f; }
#pragma unroll
  for (int jj = 0; jj < 4; ++jj)
#pragma unroll
    for (int r = 0; r < 8; ++r) { mem2[jj][r] = 0.f; d2[jj][r] = 0.f; }

  v8f zero = {0.f, 0.f, 0.f, 0.f, 0.f, 0.f, 0.f, 0.f};

  // Prologue: async-copy din1 slice for t=0 into dbuf[0].
  // Wave w copies bytes [w*4096, w*4096+4096) => rows m1 in {2w, 2w+1}, which is
  // exactly the region its own threads consume -> wave-local asynccnt wait suffices.
  {
    const char* src = (const char*)(din1 + (size_t)b0 * H1_) + w * 4096 + lane * 16;
    unsigned    dst = (unsigned)(uintptr_t)((char*)&dbuf[0][0] + w * 4096 + lane * 16);
#pragma unroll
    for (int i = 0; i < 8; ++i)
      async_copy16(dst + i * 512, (unsigned long long)(src + i * 512));
  }
  __syncthreads();

  for (int t = 0; t < T_; ++t) {
    // ---- layer 1 state update (registers), spikes -> LDS in A-frag order ----
    wait_async0();
    const float* dp = &dbuf[t & 1][m1 * H1_ + hbase];
    _Float16 lo[16], hi[16];    // lane m1 gets [h0..7,h16..23]; lane m1+16 gets [h8..15,h24..31]
#pragma unroll
    for (int i = 0; i < 32; ++i) {
      float be = beta1s[hbase + i], al = alpha1s[hbase + i];
      float di = dp[i];
      d1[i] = be * d1[i] + (1.0f - be) * di;
      float mv = al * mem1[i] + (1.0f - al) * d1[i];
      float sp = (mv > 1.0f) ? 1.0f : 0.0f;
      mem1[i] = mv * (1.0f - sp);
      _Float16 sh = (_Float16)sp;
      int g = (i & 15) >> 3;                  // lane group for this k
      int base = i - 8 * g;                   // in [0,8) or [16,24)
      int e = (base < 8) ? base : (base - 8);
      if (g == 0) lo[e] = sh; else hi[e] = sh;
    }
    *(v16h*)(s1p + ((kt1 * 32 + m1) << 4))      = *(v16h*)lo;
    *(v16h*)(s1p + ((kt1 * 32 + m1 + 16) << 4)) = *(v16h*)hi;
    __syncthreads();

    // ---- issue async copy of next din1 slice (overlaps both GEMMs) ----
    if (t + 1 < T_) {
      const char* src = (const char*)(din1 + (size_t)((t + 1) * 64 + b0) * H1_) + w * 4096 + lane * 16;
      unsigned    dst = (unsigned)(uintptr_t)((char*)&dbuf[(t + 1) & 1][0] + w * 4096 + lane * 16);
#pragma unroll
      for (int i = 0; i < 8; ++i)
        async_copy16(dst + i * 512, (unsigned long long)(src + i * 512));
    }

    // ---- layer 2 GEMM: s1[16x512] @ W2^T, wave w -> N-tiles w*4..w*4+3 ----
    v8f c2[4];
#pragma unroll
    for (int jj = 0; jj < 4; ++jj) c2[jj] = zero;
    for (int kt = 0; kt < 16; ++kt) {
      v16h a = *(const v16h*)(s1p + ((kt * 32 + lane) << 4));
#pragma unroll
      for (int jj = 0; jj < 4; ++jj) {
        v16h bfr = *(const v16h*)(w2p + ((size_t)((w * 4 + jj) * 16 + kt) * 32 + lane) * 16);
        c2[jj] = __builtin_amdgcn_wmma_f32_16x16x32_f16(
            false, a, false, bfr, (short)0, c2[jj], false, false);
      }
    }
    // ---- layer 2 state update; spikes -> LDS in A-frag order ----
#pragma unroll
    for (int jj = 0; jj < 4; ++jj) {
      int n   = (w * 4 + jj) * 16 + ln;
      int kt2 = n >> 5;
      int kk  = n & 31;
      int g2  = ((kk & 15) >= 8) ? 1 : 0;
      int bse = kk - 8 * g2;
      int e2  = (bse < 8) ? bse : (bse - 8);
#pragma unroll
      for (int r = 0; r < 8; ++r) {
        int m = r + 8 * lg;
        float di = c2[jj][r] + b2r[jj];
        d2[jj][r] = beta2r[jj] * d2[jj][r] + (1.0f - beta2r[jj]) * di;
        float mv = alpha2r[jj] * mem2[jj][r] + (1.0f - alpha2r[jj]) * d2[jj][r];
        float sp = (mv > 1.0f) ? 1.0f : 0.0f;
        mem2[jj][r] = mv * (1.0f - sp);
        s2p[((kt2 * 32 + m + 16 * g2) << 4) + e2] = (_Float16)sp;
      }
    }
    __syncthreads();

    // ---- output GEMM: s2[16x512] @ Wo^T, wave w -> N-tile w ----
    v8f co = zero;
    for (int kt = 0; kt < 16; ++kt) {
      v16h a   = *(const v16h*)(s2p + ((kt * 32 + lane) << 4));
      v16h bfr = *(const v16h*)(wop + ((size_t)(w * 16 + kt) * 32 + lane) * 16);
      co = __builtin_amdgcn_wmma_f32_16x16x32_f16(
          false, a, false, bfr, (short)0, co, false, false);
    }
#pragma unroll
    for (int r = 0; r < 8; ++r) {
      int m = r + 8 * lg;
      out[((size_t)(b0 + m) * T_ + t) * O_ + nO] = sigmoidf_(co[r] + bo_r);
    }
    __syncthreads();
  }
}

// ---------------------------------------------------------------------------
extern "C" void kernel_launch(void* const* d_in, const int* in_sizes, int n_in,
                              void* d_out, int out_size, void* d_ws, size_t ws_size,
                              hipStream_t stream) {
  const float* x      = (const float*)d_in[0];
  const float* W1     = (const float*)d_in[1];
  const float* b1     = (const float*)d_in[2];
  const float* tau_m1 = (const float*)d_in[3];
  const float* tau_n1 = (const float*)d_in[4];
  const float* W2     = (const float*)d_in[5];
  const float* b2     = (const float*)d_in[6];
  const float* tau_m2 = (const float*)d_in[7];
  const float* tau_n2 = (const float*)d_in[8];
  const float* Wo     = (const float*)d_in[9];
  const float* bo     = (const float*)d_in[10];
  float* out = (float*)d_out;

  char* ws = (char*)d_ws;
  _Float16* xp  = (_Float16*)ws;                                   // 33,554,432 B
  _Float16* w1p = (_Float16*)(ws + 33554432);                      //    262,144 B
  _Float16* w2p = (_Float16*)(ws + 33554432 + 262144);             //    524,288 B
  _Float16* wop = (_Float16*)(ws + 33554432 + 262144 + 524288);    //    131,072 B
  float*    din1 = (float*)(ws + 33554432 + 262144 + 524288 + 131072); // 134,217,728 B

  pack_x_kernel<<<65536, 256, 0, stream>>>(x, xp);
  pack_w_kernel<<<1792, 256, 0, stream>>>(W1, W2, Wo, w1p, w2p, wop);
  gemm1_kernel<<<1024, 256, 0, stream>>>(xp, w1p, b1, din1);
  scan_kernel<<<4, 256, 0, stream>>>(din1, w2p, wop, b2, bo,
                                     tau_m1, tau_n1, tau_m2, tau_n2, out);
}
